// MultiHeadSelfAttention_15573551416080
// MI455X (gfx1250) — matrix operand order repaired and checked by reading, not compiled, not run on hardware
//
#include <hip/hip_runtime.h>
#include <stdint.h>

#define DMODEL 2048
#define NHEADS 16
#define DK     128
#define BATCH  4
#define SEQ    2048
#define BS     (BATCH*SEQ)

typedef __attribute__((ext_vector_type(16))) __bf16 v16bf;
typedef __attribute__((ext_vector_type(8)))  float  v8f;
typedef __attribute__((ext_vector_type(4)))  uint32_t v4u;
typedef __attribute__((ext_vector_type(8)))  int      v8i_;
typedef __attribute__((ext_vector_type(4)))  int      v4i_;

union BF16x16 { v16bf v; uint32_t u[8]; };

// A-matrix (16x32 bf16) K index for VGPR j, lane L (CDNA5 ISA 16-bit A layout)
__device__ __forceinline__ int kApat(int j, int L) {
    return ((j & 4) << 2) + ((L & 16) >> 1) + ((j & 3) << 1);
}
// B-matrix (32x16 bf16) K index for VGPR j, lane L (lane%16 = N, K pairs per VGPR, same K order as A)
__device__ __forceinline__ int kBpat(int j, int L) {
    return ((j & 4) << 2) + ((L & 16) >> 1) + ((j & 3) << 1);
}

__device__ __forceinline__ uint32_t bf16_1(float a) {
    uint32_t u = __float_as_uint(a);
    return (u + 0x7FFFu + ((u >> 16) & 1u)) >> 16;   // RNE
}
__device__ __forceinline__ uint32_t pk_bf16(float lo, float hi) {
    return (bf16_1(lo) & 0xFFFFu) | (bf16_1(hi) << 16);
}
__device__ __forceinline__ v8f wmma_bf16(v16bf a, v16bf b, v8f c) {
    return __builtin_amdgcn_wmma_f32_16x16x32_bf16(false, a, false, b, (short)0, c, false, false);
}

// ---------------------------------------------------------------------------
// Tensor Data Mover: load a 2D tile (tile1 rows x tile0 elements, bf16) from
// global (row stride stride0 elements) into LDS at byte offset lds_off.
// Descriptor packing per CDNA5 ISA.
// (6-arg builtin form: g0, g1, g2, g3, g4-spare, cpol)
// ---------------------------------------------------------------------------
__device__ __forceinline__ void tdm_load_2d(uint32_t lds_off, const void* gptr,
                                            uint32_t tile0, uint32_t tile1,
                                            uint32_t stride0)
{
    uint64_t ga = (uint64_t)(uintptr_t)gptr;
    v4u g0;
    g0[0] = 1u;                                   // count=1 (valid user descriptor)
    g0[1] = lds_off;                              // LDS byte address
    g0[2] = (uint32_t)ga;                         // global_addr[31:0]
    g0[3] = (uint32_t)(ga >> 32) | (2u << 30);    // global_addr[56:32] | type=2
    const uint32_t td0 = 1u << 20, td1 = 1u << 20; // large tensor dims: no OOB clip
    v8i_ g1;
    g1[0] = (int)(1u << 16);                      // data_size = 1 (2 bytes)
    g1[1] = (int)((td0 & 0xFFFFu) << 16);                         // tensor_dim0 lo
    g1[2] = (int)((td0 >> 16) | ((td1 & 0xFFFFu) << 16));         // td0 hi | td1 lo
    g1[3] = (int)(((td1 >> 16) & 0xFFFFu) | (tile0 << 16));       // td1 hi | tile_dim0
    g1[4] = (int)(tile1 & 0xFFFFu);                               // tile_dim1
    g1[5] = (int)stride0;                                         // tensor_dim0_stride
    g1[6] = 0;
    g1[7] = 0;
    v4i_ z4 = (v4i_)0;
    v8i_ z8 = (v8i_)0;
    __builtin_amdgcn_tensor_load_to_lds(g0, g1, z4, z4, z8, 0);
}

// ---------------------------------------------------------------------------
// Bulk fp32 -> bf16 conversion, 4 elements per thread (b128 load, b64 store).
// ---------------------------------------------------------------------------
__global__ __launch_bounds__(256) void convert_bf16(const float* __restrict__ src,
                                                    uint16_t* __restrict__ dst, int n4)
{
    int tid = blockIdx.x * blockDim.x + threadIdx.x;
    if (tid >= n4) return;
    float4 f = ((const float4*)src)[tid];
    uint2 o;
    o.x = pk_bf16(f.x, f.y);
    o.y = pk_bf16(f.z, f.w);
    ((uint2*)dst)[tid] = o;
}

// ---------------------------------------------------------------------------
// Y[M,N] = A[M,K] * W[N,K]^T ; A,W bf16 (pre-converted), Y bf16 or fp32,
// optionally stored transposed (Y[N,M]) for the V matrix.
// One wave computes a 32x64 strip (2 A x 4 B = 8 WMMAs per 32-K step).
// K-loop unrolled x2 with two register buffer sets (software pipelining).
// ---------------------------------------------------------------------------
template<bool OUT_BF16, bool OUT_T>
__global__ __launch_bounds__(128) void gemm_bbT(const uint16_t* __restrict__ A,
                                                const uint16_t* __restrict__ W,
                                                void* __restrict__ Y,
                                                int M, int N, int K)
{
    int L   = threadIdx.x & 31;
    int wid = threadIdx.x >> 5;
    int gw  = blockIdx.x * 4 + wid;
    int nt64 = N >> 6;
    int mt  = gw / nt64;
    int n64 = gw - mt * nt64;
    if (mt * 32 >= M) return;
    int m0 = mt * 32, n0 = n64 * 64;
    int lrow = L & 15;

    v8f acc[2][4];
    #pragma unroll
    for (int r = 0; r < 2; r++)
        #pragma unroll
        for (int t = 0; t < 4; t++) acc[r][t] = (v8f)0.0f;

    const uint16_t* ar0 = A + (size_t)(m0 + lrow) * K;
    const uint16_t* ar1 = ar0 + (size_t)16 * K;

    auto loadAB = [&](BF16x16& A0, BF16x16& A1, BF16x16* B, int k) {
        #pragma unroll
        for (int j = 0; j < 8; j++) {
            int ka = k + kApat(j, L);
            A0.u[j] = *(const uint32_t*)(ar0 + ka);
            A1.u[j] = *(const uint32_t*)(ar1 + ka);
        }
        #pragma unroll
        for (int t = 0; t < 4; t++) {
            const uint16_t* wr = W + (size_t)(n0 + 16 * t + lrow) * K + k;
            #pragma unroll
            for (int j = 0; j < 8; j++)
                B[t].u[j] = *(const uint32_t*)(wr + kBpat(j, L));
        }
    };

    BF16x16 aX0, aX1, bX[4];
    BF16x16 aY0, aY1, bY[4];
    loadAB(aX0, aX1, bX, 0);

    for (int k0 = 0; k0 < K; k0 += 64) {
        loadAB(aY0, aY1, bY, k0 + 32);           // prefetch next half-step
        #pragma unroll
        for (int t = 0; t < 4; t++) {
            acc[0][t] = wmma_bf16(aX0.v, bX[t].v, acc[0][t]);
            acc[1][t] = wmma_bf16(aX1.v, bX[t].v, acc[1][t]);
        }
        int k2 = (k0 + 64 < K) ? k0 + 64 : 0;    // last prefetch wraps (unused)
        loadAB(aX0, aX1, bX, k2);
        #pragma unroll
        for (int t = 0; t < 4; t++) {
            acc[0][t] = wmma_bf16(aY0.v, bY[t].v, acc[0][t]);
            acc[1][t] = wmma_bf16(aY1.v, bY[t].v, acc[1][t]);
        }
    }

    #pragma unroll
    for (int r = 0; r < 2; r++) {
        int rbase = m0 + 16 * r + ((L & 16) >> 1);   // rows j / j+8 across lane halves
        #pragma unroll
        for (int t = 0; t < 4; t++) {
            #pragma unroll
            for (int j = 0; j < 8; j++) {
                int row = rbase + j;
                int col = n0 + 16 * t + lrow;
                size_t idx = OUT_T ? ((size_t)col * M + row) : ((size_t)row * N + col);
                if constexpr (OUT_BF16) ((uint16_t*)Y)[idx] = (uint16_t)bf16_1(acc[r][t][j]);
                else                    ((float*)Y)[idx]    = acc[r][t][j];
            }
        }
    }
}

// ---------------------------------------------------------------------------
// RoPE applied in-place to bf16 Q and K (adjacent even/odd dims = one dword).
// Q is additionally pre-scaled by 1/sqrt(d_k) so attention needs no score scale.
// ---------------------------------------------------------------------------
__global__ __launch_bounds__(256) void rope_kernel(uint32_t* __restrict__ Q,
                                                   uint32_t* __restrict__ K,
                                                   const int* __restrict__ tp)
{
    int tid = blockIdx.x * blockDim.x + threadIdx.x;   // 0 .. BS*1024-1
    int row = tid >> 10;
    int i   = tid & 63;
    float pos  = (float)tp[row];
    float freq = __expf(-(float)(2 * i) * (1.0f / (float)DK) * __logf(10000.0f));
    float sn, cs;
    __sincosf(pos * freq, &sn, &cs);
    const float sm = 0.08838834764831845f;   // 1/sqrt(128)

    uint32_t q  = Q[tid];
    float q1 = __uint_as_float(q << 16);
    float q2 = __uint_as_float(q & 0xFFFF0000u);
    Q[tid] = pk_bf16((q1 * cs - q2 * sn) * sm, (q1 * sn + q2 * cs) * sm);

    uint32_t k  = K[tid];
    float k1 = __uint_as_float(k << 16);
    float k2 = __uint_as_float(k & 0xFFFF0000u);
    K[tid] = pk_bf16(k1 * cs - k2 * sn, k1 * sn + k2 * cs);
}

// ---------------------------------------------------------------------------
// Causal flash attention, block-cooperative version.
// A block's 4 waves handle 4 consecutive q-tiles of the SAME (b,h), so they
// share every 32-token K/V chunk. Chunks are staged in LDS by the Tensor Data
// Mover (one tensor_load_to_lds per tile, double-buffered): wave 0 issues the
// TDM for chunk c+1, everyone computes chunk c from LDS, then
// s_wait_tensorcnt + workgroup barrier, swap.
// LDS: K 2x8KB | V 2x8KB | P-transpose scratch 4x2KB = 40 KB.
// ---------------------------------------------------------------------------
__global__ __launch_bounds__(128) void flash_attn(const uint16_t* __restrict__ Qb,
                                                  const uint16_t* __restrict__ Kb,
                                                  const uint16_t* __restrict__ Vt,
                                                  uint16_t* __restrict__ Ob)
{
    __shared__ __align__(16) uint8_t smem[40960];
    int L   = threadIdx.x & 31;
    int wid = threadIdx.x >> 5;
    int blk = blockIdx.x;                 // 0 .. 2047
    int qg  = blk & 31;                   // SEQ/64 groups
    int h   = (blk >> 5) & 15;
    int b   = blk >> 9;
    int qbase = qg << 6;
    int q0  = qbase + wid * 16;           // this wave's q-tile
    int lrow = L & 15;
    int h8   = (L & 16) >> 1;             // row offset 0 / 8 per lane half
    float* lp = (float*)(smem + 32768 + wid * 2048);

    // Q tile as 4 A-operands (d_k = 4 x 32)
    BF16x16 qa[4];
    const uint16_t* qrow = Qb + (size_t)(b * SEQ + q0 + lrow) * DMODEL + h * DK;
    #pragma unroll
    for (int d = 0; d < 4; d++)
        #pragma unroll
        for (int j = 0; j < 8; j++)
            qa[d].u[j] = *(const uint32_t*)(qrow + d * 32 + kApat(j, L));

    // constant all-ones B tile (bf16 1.0 packed), lives purely in registers
    BF16x16 onesb;
    #pragma unroll
    for (int j = 0; j < 8; j++) onesb.u[j] = 0x3F803F80u;

    v8f o[8], ol;
    #pragma unroll
    for (int t = 0; t < 8; t++) o[t] = (v8f)0.0f;
    ol = (v8f)0.0f;
    float mrow[8];
    #pragma unroll
    for (int j = 0; j < 8; j++) mrow[j] = -1e30f;

    const uint16_t* gK = Kb + (size_t)(b * SEQ) * DMODEL + h * DK;  // [token][dk]
    const uint16_t* gV = Vt + (size_t)(h * DK) * BS + b * SEQ;      // [dk][token]
    int nchunks = qbase / 32 + 2;          // covers rows up to qbase+63

    // prologue: stage chunk 0 into buffer 0
    if (wid == 0) {
        tdm_load_2d(0,     gK, 128, 32, DMODEL);   // K tile: 32 rows x 128
        tdm_load_2d(16384, gV, 32, 128, BS);       // V tile: 128 rows x 32
        __builtin_amdgcn_s_wait_tensorcnt(0);
    }
    __syncthreads();

    int buf = 0;
    for (int c = 0; c < nchunks; c++) {
        int kk0 = c * 32;
        if (wid == 0 && c + 1 < nchunks) {         // prefetch chunk c+1
            tdm_load_2d((buf ^ 1) * 8192,         gK + (size_t)(kk0 + 32) * DMODEL, 128, 32, DMODEL);
            tdm_load_2d(16384 + (buf ^ 1) * 8192, gV + (kk0 + 32),                   32, 128, BS);
        }
        const uint16_t* kt = (const uint16_t*)(smem + buf * 8192);          // [32][128]
        const uint16_t* vt = (const uint16_t*)(smem + 16384 + buf * 8192);  // [128][32]

        // ---- scores S[16q x 32k] = Q . K^T (Q pre-scaled by 1/sqrt(dk)) ----
        v8f s0 = (v8f)0.0f, s1 = (v8f)0.0f;
        #pragma unroll
        for (int d = 0; d < 4; d++) {
            BF16x16 kb;
            #pragma unroll
            for (int j = 0; j < 8; j++)
                kb.u[j] = *(const uint32_t*)(kt + (size_t)lrow * 128 + d * 32 + kBpat(j, L));
            s0 = wmma_bf16(qa[d].v, kb.v, s0);
        }
        #pragma unroll
        for (int d = 0; d < 4; d++) {
            BF16x16 kb;
            #pragma unroll
            for (int j = 0; j < 8; j++)
                kb.u[j] = *(const uint32_t*)(kt + (size_t)(16 + lrow) * 128 + d * 32 + kBpat(j, L));
            s1 = wmma_bf16(qa[d].v, kb.v, s1);
        }

        // ---- causal mask: boundary / beyond chunks only (uniform branch) ----
        if (kk0 + 31 > q0) {
            #pragma unroll
            for (int j = 0; j < 8; j++) {
                int qg2 = q0 + j + h8;
                int kg0 = kk0 + lrow;
                s0[j] = (kg0      <= qg2) ? s0[j] : -3.0e38f;
                s1[j] = (kg0 + 16 <= qg2) ? s1[j] : -3.0e38f;
            }
        }

        // ---- online softmax: row max reduction + rescale (sum via WMMA) ----
        #pragma unroll
        for (int j = 0; j < 8; j++) {
            float v = fmaxf(s0[j], s1[j]);
            #pragma unroll
            for (int m = 8; m >= 1; m >>= 1) v = fmaxf(v, __shfl_xor(v, m, 32));
            float mnew = fmaxf(mrow[j], v);
            float corr = __expf(mrow[j] - mnew);
            mrow[j] = mnew;
            s0[j] = __expf(s0[j] - mnew);
            s1[j] = __expf(s1[j] - mnew);
            #pragma unroll
            for (int t = 0; t < 8; t++) o[t][j] *= corr;
            ol[j] *= corr;
        }

        // ---- C-layout -> A-layout transpose of P through per-wave LDS ----
        #pragma unroll
        for (int j = 0; j < 8; j++) {
            int row = j + h8;
            lp[row * 32 + lrow]      = s0[j];
            lp[row * 32 + 16 + lrow] = s1[j];
        }
        asm volatile("s_wait_dscnt 0" ::: "memory");
        BF16x16 pa;
        #pragma unroll
        for (int j = 0; j < 8; j++) {
            int k = kApat(j, L);
            pa.u[j] = pk_bf16(lp[lrow * 32 + k], lp[lrow * 32 + k + 1]);
        }
        asm volatile("s_wait_dscnt 0" ::: "memory");

        // ---- O += P . V (8 tiles from LDS) ; l += P . 1 (no loads) ----
        #pragma unroll
        for (int t = 0; t < 8; t++) {
            BF16x16 vb;
            #pragma unroll
            for (int j = 0; j < 8; j++)
                vb.u[j] = *(const uint32_t*)(vt + (size_t)(t * 16 + lrow) * 32 + kBpat(j, L));
            o[t] = wmma_bf16(pa.v, vb.v, o[t]);
        }
        ol = wmma_bf16(pa.v, onesb.v, ol);

        if (wid == 0) __builtin_amdgcn_s_wait_tensorcnt(0);
        __syncthreads();
        buf ^= 1;
    }

    // ---- normalize + store bf16 (l is lane-replicated in ol) ----
    uint16_t* obase = Ob + (size_t)(b * SEQ + q0 + h8) * DMODEL + h * DK + lrow;
    #pragma unroll
    for (int j = 0; j < 8; j++) {
        float inv = 1.0f / ol[j];
        #pragma unroll
        for (int t = 0; t < 8; t++)
            obase[(size_t)j * DMODEL + t * 16] = (uint16_t)bf16_1(o[t][j] * inv);
    }
}

// ---------------------------------------------------------------------------
extern "C" void kernel_launch(void* const* d_in, const int* in_sizes, int n_in,
                              void* d_out, int out_size, void* d_ws, size_t ws_size,
                              hipStream_t stream)
{
    (void)in_sizes; (void)n_in; (void)out_size; (void)ws_size;
    const float* x  = (const float*)d_in[0];
    const float* Wq = (const float*)d_in[1];
    const float* Wk = (const float*)d_in[2];
    const float* Wv = (const float*)d_in[3];
    const float* Wo = (const float*)d_in[4];
    const int*   tp = (const int*)d_in[5];
    float* out = (float*)d_out;

    // Workspace layout (bf16): Q,K,Vt,O (32 MB each) + x (32 MB) + 4 weights (8 MB each)
    uint16_t* Qb  = (uint16_t*)d_ws;
    uint16_t* Kb  = Qb  + (size_t)BS * DMODEL;
    uint16_t* Vtb = Kb  + (size_t)BS * DMODEL;
    uint16_t* Ob  = Vtb + (size_t)BS * DMODEL;
    uint16_t* xb  = Ob  + (size_t)BS * DMODEL;
    uint16_t* wqb = xb  + (size_t)BS * DMODEL;
    uint16_t* wkb = wqb + (size_t)DMODEL * DMODEL;
    uint16_t* wvb = wkb + (size_t)DMODEL * DMODEL;
    uint16_t* wob = wvb + (size_t)DMODEL * DMODEL;

    const int xN4 = BS * DMODEL / 4;          // 4,194,304
    const int wN4 = DMODEL * DMODEL / 4;      // 1,048,576
    convert_bf16<<<xN4 / 256, 256, 0, stream>>>(x,  xb,  xN4);
    convert_bf16<<<wN4 / 256, 256, 0, stream>>>(Wq, wqb, wN4);
    convert_bf16<<<wN4 / 256, 256, 0, stream>>>(Wk, wkb, wN4);
    convert_bf16<<<wN4 / 256, 256, 0, stream>>>(Wv, wvb, wN4);
    convert_bf16<<<wN4 / 256, 256, 0, stream>>>(Wo, wob, wN4);

    const int gemmBlocks = (BS / 32) * (DMODEL / 64) / 4;   // 2048 blocks x 4 waves

    gemm_bbT<true,  false><<<gemmBlocks, 128, 0, stream>>>(xb, wqb, (void*)Qb,  BS, DMODEL, DMODEL);
    gemm_bbT<true,  false><<<gemmBlocks, 128, 0, stream>>>(xb, wkb, (void*)Kb,  BS, DMODEL, DMODEL);
    gemm_bbT<true,  true ><<<gemmBlocks, 128, 0, stream>>>(xb, wvb, (void*)Vtb, BS, DMODEL, DMODEL);

    rope_kernel<<<(BS * (DMODEL / 2)) / 256, 256, 0, stream>>>((uint32_t*)Qb, (uint32_t*)Kb, tp);

    flash_attn<<<BATCH * NHEADS * (SEQ / 64), 128, 0, stream>>>(Qb, Kb, Vtb, Ob);

    gemm_bbT<false, false><<<gemmBlocks, 128, 0, stream>>>(Ob, wob, (void*)out, BS, DMODEL, DMODEL);
}
